// SingleHead_91087666414083
// MI455X (gfx1250) — matrix operand order repaired and checked by reading, not compiled, not run on hardware
//
#include <hip/hip_runtime.h>
#include <hip/hip_bf16.h>

typedef __bf16 bf16;
typedef __attribute__((ext_vector_type(16))) bf16  v16bf;
typedef __attribute__((ext_vector_type(8)))  bf16  v8bf;
typedef __attribute__((ext_vector_type(4)))  bf16  v4bf;
typedef __attribute__((ext_vector_type(8)))  float v8f;
typedef __attribute__((ext_vector_type(4)))  unsigned int v4u;
typedef __attribute__((ext_vector_type(8)))  int v8i;
typedef __attribute__((ext_vector_type(4)))  int v4i;

#define B_ 256
#define T_ 256
#define C_ 512
#define H_ 64

static constexpr long XB_ELEMS = (long)B_ * T_ * C_;   // 33,554,432 bf16
static constexpr long WT_ELEMS = (long)3 * H_ * C_;    // 98,304 bf16
static constexpr long QK_ELEMS = (long)B_ * T_ * H_;   // 4,194,304 bf16

__device__ __forceinline__ v8f wmma_bf16(v16bf a, v16bf b, v8f c) {
  return __builtin_amdgcn_wmma_f32_16x16x32_bf16(false, a, false, b, (short)0, c,
                                                 false, false);
}

__device__ __forceinline__ v8f v8f_zero() {
  v8f z;
  #pragma unroll
  for (int i = 0; i < 8; ++i) z[i] = 0.0f;
  return z;
}

// ---------------- Kernel 1: x fp32 -> bf16 ----------------
__global__ void prep_x(const float* __restrict__ x, bf16* __restrict__ xb, long n4) {
  long i = (long)blockIdx.x * blockDim.x + threadIdx.x;
  if (i >= n4) return;
  float4 f = ((const float4*)x)[i];
  v4bf o;
  o[0] = (bf16)f.x; o[1] = (bf16)f.y; o[2] = (bf16)f.z; o[3] = (bf16)f.w;
  ((v4bf*)xb)[i] = o;
}

// ---------------- Kernel 2: W -> W^T bf16 (slots: 0=q, 1=k, 2=v) ----------------
__global__ void prep_w(const float* __restrict__ Wq, const float* __restrict__ Wk,
                       const float* __restrict__ Wv, bf16* __restrict__ wt) {
  int i = blockIdx.x * blockDim.x + threadIdx.x;
  if (i >= C_ * H_) return;
  int k = i / H_;
  int h = i % H_;
  wt[0 * H_ * C_ + h * C_ + k] = (bf16)Wq[i];
  wt[1 * H_ * C_ + h * C_ + k] = (bf16)Wk[i];
  wt[2 * H_ * C_ + h * C_ + k] = (bf16)Wv[i];
}

// ---------------- Kernel 3: q/k/v projections via TDM + WMMA ----------------
// 4 waves / block; each wave owns a 16-row slab of x (16x512 bf16 = 16KB),
// staged into LDS by one tensor_load_to_lds. nt is the outer loop so only
// 3 accumulators are live, letting the scheduler prefetch B operands.
__global__ void __launch_bounds__(128) proj_qkv(
    const bf16* __restrict__ xb, const bf16* __restrict__ wt,
    const float* __restrict__ biasq, const float* __restrict__ biask,
    const float* __restrict__ biasv,
    bf16* __restrict__ q, bf16* __restrict__ k, bf16* __restrict__ vT) {
  __shared__ bf16 aslab[4][16][512];  // 64KB

  const int lane = threadIdx.x & 31;
  const int wave = __builtin_amdgcn_readfirstlane((int)(threadIdx.x >> 5));
  const int half = lane >> 4;
  const int ln   = lane & 15;
  const long r0  = ((long)blockIdx.x * 4 + wave) * 16;

  // ---- TDM: stage xb[r0*512 .. r0*512+8192) into aslab[wave] ----
  {
    const unsigned lds_off = (unsigned)(unsigned long long)(size_t)&aslab[wave][0][0];
    const unsigned long long ga = (unsigned long long)(size_t)(xb + r0 * C_);
    // D# group0: count=1 | lds_addr | global_addr[56:0] | type=2
    v4u g0;
    g0[0] = 1u;                                         // count=1, user mode
    g0[1] = lds_off;                                    // LDS byte address
    g0[2] = (unsigned)(ga & 0xFFFFFFFFu);               // global_addr[31:0]
    g0[3] = (unsigned)((ga >> 32) & 0x01FFFFFFu) | (2u << 30);  // addr[56:32], type=2
    // D# group1: wg_mask=0, data_size=2B, tile 512x16, tensor_dim0=512,
    // tensor_dim1=65536, dim0 stride=512
    v8i g1;
    g1[0] = (int)(1u << 16);                 // data_size = 1 (2 bytes)
    g1[1] = (int)(512u << 16);               // tensor_dim0[15:0] = 512
    g1[2] = 0;                               // tensor_dim0[31:16]=0, tensor_dim1[15:0]=0
    g1[3] = (int)((512u << 16) | 1u);        // tensor_dim1[31:16]=1 (=65536), tile_dim0=512
    g1[4] = 16;                              // tile_dim1 = 16, tile_dim2 = 0
    g1[5] = 512;                             // tensor_dim0_stride[31:0] = 512
    g1[6] = 0;                               // stride0 hi, tensor_dim1_stride lo
    g1[7] = 0;
    v4i g2; g2[0] = 0; g2[1] = 0; g2[2] = 0; g2[3] = 0;
    v4i g3; g3[0] = 0; g3[1] = 0; g3[2] = 0; g3[3] = 0;
    v8i g4;
    #pragma unroll
    for (int i = 0; i < 8; ++i) g4[i] = 0;
    __builtin_amdgcn_tensor_load_to_lds(g0, g1, g2, g3, g4, 0);
  }
  __builtin_amdgcn_s_wait_tensorcnt(0);
  asm volatile("" ::: "memory");

  const bf16* wtq = wt;
  const bf16* wtk = wt + (long)H_ * C_;
  const bf16* wtv = wt + 2L * H_ * C_;

  #pragma unroll
  for (int nt = 0; nt < 4; ++nt) {
    v8f aq = v8f_zero(), ak = v8f_zero(), av = v8f_zero();
    const int h = nt * 16 + ln;
    const long brow = (long)h * C_ + half * 16;

    // prefetch kc=0 B operands
    v16bf bq = *(const v16bf*)(wtq + brow);
    v16bf bk = *(const v16bf*)(wtk + brow);
    v16bf bv = *(const v16bf*)(wtv + brow);

    for (int kc = 0; kc < C_ / 32; ++kc) {
      // A operand from LDS slab: row = l%16, K contiguous (k = 16*half + s)
      v8bf alo = *(const v8bf*)&aslab[wave][ln][kc * 32 + half * 16];
      v8bf ahi = *(const v8bf*)&aslab[wave][ln][kc * 32 + half * 16 + 8];
      v16bf a = __builtin_shufflevector(alo, ahi,
                                        0,1,2,3,4,5,6,7,8,9,10,11,12,13,14,15);
      // prefetch next kc's B operands (clamped wrap on last iter; harmless)
      const long nb = brow + (long)(((kc + 1) & 15) * 32);
      v16bf nbq = *(const v16bf*)(wtq + nb);
      v16bf nbk = *(const v16bf*)(wtk + nb);
      v16bf nbv = *(const v16bf*)(wtv + nb);

      aq = wmma_bf16(a, bq, aq);
      ak = wmma_bf16(a, bk, ak);
      av = wmma_bf16(a, bv, av);

      bq = nbq; bk = nbk; bv = nbv;
    }

    // Bias add + store. C layout: lane l holds (row=(l/16)*8+j, col=l%16) in VGPR j.
    const float bqv = biasq[h], bkv = biask[h], bvv = biasv[h];
    #pragma unroll
    for (int j = 0; j < 8; ++j) {
      const long row = r0 + half * 8 + j;
      q[row * H_ + h] = (bf16)(aq[j] + bqv);
      k[row * H_ + h] = (bf16)(ak[j] + bkv);
      const long b = row / T_;
      const long t = row % T_;
      vT[(b * H_ + h) * T_ + t] = (bf16)(av[j] + bvv);
    }
  }
}

// ---------------- Kernel 4: causal flash attention via WMMA ----------------
// One wave per 16-query tile; key blocks of 32; online softmax.
__global__ void __launch_bounds__(256) attn(
    const bf16* __restrict__ q, const bf16* __restrict__ kbuf,
    const bf16* __restrict__ vT, float* __restrict__ out) {
  __shared__ bf16 pb[8][16][32];  // per-wave P tile re-layout buffer (C-format -> A-format)

  const int lane = threadIdx.x & 31;
  const int wave = threadIdx.x >> 5;
  const int half = lane >> 4;
  const int ln   = lane & 15;
  const int g    = blockIdx.x * 8 + wave;
  const int b    = g >> 4;
  const int t0   = (g & 15) << 4;

  // Q A-operands for the two K-chunks (h 0..31, 32..63)
  const bf16* qrow = q + ((long)b * T_ + t0 + ln) * H_;
  v16bf aq0 = __builtin_shufflevector(*(const v8bf*)(qrow + half * 16),
                                      *(const v8bf*)(qrow + 8 + half * 16),
                                      0,1,2,3,4,5,6,7,8,9,10,11,12,13,14,15);
  v16bf aq1 = __builtin_shufflevector(*(const v8bf*)(qrow + 32 + half * 16),
                                      *(const v8bf*)(qrow + 40 + half * 16),
                                      0,1,2,3,4,5,6,7,8,9,10,11,12,13,14,15);

  float m_r[8], l_r[8];
  #pragma unroll
  for (int j = 0; j < 8; ++j) { m_r[j] = -__builtin_inff(); l_r[j] = 0.0f; }
  v8f acc[4];
  #pragma unroll
  for (int nt = 0; nt < 4; ++nt) acc[nt] = v8f_zero();

  const float scale = 0.125f;  // 1/sqrt(64)
  const int nkb = ((t0 + 15) >> 5) + 1;

  for (int kb2 = 0; kb2 < nkb; ++kb2) {
    const int s_base = kb2 * 32;

    // scores: two 16x16 tiles (s-groups), K=64 via two chunks each
    v8f s0 = v8f_zero(), s1 = v8f_zero();
    {
      const bf16* kr0 = kbuf + ((long)b * T_ + s_base + ln) * H_;
      const bf16* kr1 = kr0 + 16 * H_;
      v16bf bk00 = *(const v16bf*)(kr0 + half * 16);
      v16bf bk01 = *(const v16bf*)(kr0 + 32 + half * 16);
      v16bf bk10 = *(const v16bf*)(kr1 + half * 16);
      v16bf bk11 = *(const v16bf*)(kr1 + 32 + half * 16);
      s0 = wmma_bf16(aq0, bk00, s0);
      s0 = wmma_bf16(aq1, bk01, s0);
      s1 = wmma_bf16(aq0, bk10, s1);
      s1 = wmma_bf16(aq1, bk11, s1);
    }

    // online softmax per row j (rows of this lane: t0 + half*8 + j)
    #pragma unroll
    for (int j = 0; j < 8; ++j) {
      const int t = t0 + half * 8 + j;
      float v0 = s0[j] * scale;
      float v1 = s1[j] * scale;
      if (s_base + ln > t)       v0 = -__builtin_inff();
      if (s_base + 16 + ln > t)  v1 = -__builtin_inff();
      float mx = fmaxf(v0, v1);
      #pragma unroll
      for (int d = 1; d < 16; d <<= 1) mx = fmaxf(mx, __shfl_xor(mx, d, 32));
      const float mn = fmaxf(m_r[j], mx);
      const float al = __expf(m_r[j] - mn);
      const float p0 = __expf(v0 - mn);
      const float p1 = __expf(v1 - mn);
      float sm = p0 + p1;
      #pragma unroll
      for (int d = 1; d < 16; d <<= 1) sm += __shfl_xor(sm, d, 32);
      l_r[j] = l_r[j] * al + sm;
      m_r[j] = mn;
      #pragma unroll
      for (int nt = 0; nt < 4; ++nt) acc[nt][j] *= al;
      // stash P in LDS, row-major [16 t][32 s]
      pb[wave][half * 8 + j][ln]      = (bf16)p0;
      pb[wave][half * 8 + j][16 + ln] = (bf16)p1;
    }

    asm volatile("s_wait_dscnt 0" ::: "memory");

    // P as A-operand (16x32 bf16, A layout)
    v16bf ap = __builtin_shufflevector(*(const v8bf*)&pb[wave][ln][half * 16],
                                       *(const v8bf*)&pb[wave][ln][8 + half * 16],
                                       0,1,2,3,4,5,6,7,8,9,10,11,12,13,14,15);

    // out += P @ V ; V^T[b][h][s] gives contiguous B-operand loads
    #pragma unroll
    for (int nt = 0; nt < 4; ++nt) {
      const bf16* vr = vT + ((long)b * H_ + nt * 16 + ln) * T_ + s_base + half * 16;
      v16bf bvv = *(const v16bf*)vr;
      acc[nt] = wmma_bf16(ap, bvv, acc[nt]);
    }
  }

  // normalize + store fp32 output
  #pragma unroll
  for (int nt = 0; nt < 4; ++nt) {
    #pragma unroll
    for (int j = 0; j < 8; ++j) {
      const long row = (long)b * T_ + t0 + half * 8 + j;
      out[row * H_ + nt * 16 + ln] = acc[nt][j] / l_r[j];
    }
  }
}

extern "C" void kernel_launch(void* const* d_in, const int* in_sizes, int n_in,
                              void* d_out, int out_size, void* d_ws, size_t ws_size,
                              hipStream_t stream) {
  (void)in_sizes; (void)n_in; (void)out_size; (void)ws_size;
  const float* x  = (const float*)d_in[0];
  const float* Wk = (const float*)d_in[1];
  const float* bk = (const float*)d_in[2];
  const float* Wq = (const float*)d_in[3];
  const float* bq = (const float*)d_in[4];
  const float* Wv = (const float*)d_in[5];
  const float* bv = (const float*)d_in[6];
  float* out = (float*)d_out;

  bf16* xb  = (bf16*)d_ws;
  bf16* wt  = xb + XB_ELEMS;
  bf16* qb  = wt + WT_ELEMS;
  bf16* kbf = qb + QK_ELEMS;
  bf16* vT  = kbf + QK_ELEMS;

  const long n4 = XB_ELEMS / 4;
  prep_x<<<(int)((n4 + 255) / 256), 256, 0, stream>>>(x, xb, n4);
  prep_w<<<(C_ * H_ + 255) / 256, 256, 0, stream>>>(Wq, Wk, Wv, wt);

  const int tiles = B_ * T_ / 16;      // 4096 16-row tiles
  proj_qkv<<<tiles / 4, 128, 0, stream>>>(xb, wt, bq, bk, bv, qb, kbf, vT);
  attn<<<tiles / 8, 256, 0, stream>>>(qb, kbf, vT, out);
}
